// BahdanauAttention_41008347742752
// MI455X (gfx1250) — matrix operand order repaired and checked by reading, not compiled, not run on hardware
//
#include <hip/hip_runtime.h>
#include <hip/hip_bf16.h>
#include <stdint.h>

// Problem constants (from reference)
#define Bb 64
#define Tt 4096
#define Ee 256
#define Hh 512
#define Aa 512

typedef __attribute__((ext_vector_type(16))) __bf16        v16bf;
typedef __attribute__((ext_vector_type(8)))  float         v8f;
typedef __attribute__((ext_vector_type(4)))  unsigned int  u32x4;
typedef __attribute__((ext_vector_type(8)))  int           i32x8;
typedef __attribute__((ext_vector_type(4)))  int           i32x4;

__device__ __forceinline__ unsigned short f2bf(float x) {
  union { float f; unsigned int u; } v; v.f = x;
  unsigned int r = v.u + 0x7FFFu + ((v.u >> 16) & 1u);   // round-to-nearest-even
  return (unsigned short)(r >> 16);
}

// Native V_TANH_F32 on gfx1250; branchless fallback otherwise.
__device__ __forceinline__ float fast_tanh(float x) {
#if __has_builtin(__builtin_amdgcn_tanhf)
  return __builtin_amdgcn_tanhf(x);
#else
  const float xc = fminf(fmaxf(x, -9.0f), 9.0f);
  const float e  = __expf(2.0f * xc);
  return (e - 1.0f) / (e + 1.0f);
#endif
}

// ---------------------------------------------------------------------------
// P0: pack W1 (E x A, f32 row-major) into bf16 WMMA-B lane layout.
// Layout: idx = (((ntile*8 + kstep)*32 + lane)*16 + j) bf16 elements, where
//   lane<16 : N = ntile*16+lane,    K(j) = kstep*32 + j        (j=0..15)
//   lane>=16: N = ntile*16+lane-16, K(j) = kstep*32 + 16 + j
// ---------------------------------------------------------------------------
__global__ void pack_w1_kernel(const float* __restrict__ W1,
                               unsigned short* __restrict__ w1pack) {
  const int tid  = blockIdx.x * blockDim.x + threadIdx.x; // 0..8191
  const int lane = tid & 31;
  const int ks   = (tid >> 5) & 7;
  const int nt   = tid >> 8;
  const int col  = lane & 15;
  const int hi   = lane >> 4;
  const int a    = nt * 16 + col;
  const int e0   = ks * 32 + hi * 16;
  union { unsigned short s[16]; uint4 q[2]; } o;
#pragma unroll
  for (int j = 0; j < 16; ++j) o.s[j] = f2bf(W1[(size_t)(e0 + j) * Aa + a]);
  uint4* dst = reinterpret_cast<uint4*>(w1pack + (size_t)tid * 16);
  dst[0] = o.q[0];
  dst[1] = o.q[1];
}

// ---------------------------------------------------------------------------
// P1: ph[b][a] = hidden_state[b] @ W2[:,a] + b1[a] + b2[a]
// ---------------------------------------------------------------------------
__global__ void proj_h_kernel(const float* __restrict__ hs,
                              const float* __restrict__ W2,
                              const float* __restrict__ b1,
                              const float* __restrict__ b2,
                              float* __restrict__ ph) {
  const int b = blockIdx.x;
  const float* h = hs + (size_t)b * Hh;
  for (int a = threadIdx.x; a < Aa; a += blockDim.x) {
    float s = b1[a] + b2[a];
    for (int k = 0; k < Hh; ++k) s += h[k] * W2[(size_t)k * Aa + a];
    ph[(size_t)b * Aa + a] = s;
  }
}

// ---------------------------------------------------------------------------
// K1: fused  score[b][t] = tanh(features@W1 + ph[b]) @ Wv
// Grid: B * (T/128) blocks, 256 threads (8 waves). Wave w handles rows
// t0 + 16w .. +15. W1pack (256KB bf16) staged in LDS via one TDM descriptor.
// ---------------------------------------------------------------------------
__global__ void attn_score_kernel(const float* __restrict__ features,
                                  const unsigned short* __restrict__ w1pack,
                                  const float* __restrict__ ph,
                                  const float* __restrict__ Wv,
                                  float* __restrict__ scores) {
  extern __shared__ unsigned char smem[];   // 262144 bytes: packed W1 (bf16)
  const int tid  = threadIdx.x;
  const int wave = tid >> 5;
  const int lane = tid & 31;
  const int col  = lane & 15;
  const int hi   = lane >> 4;
  const int tiles = Tt / 128;               // 32
  const int b  = blockIdx.x / tiles;
  const int t0 = (blockIdx.x % tiles) * 128;

  // ---- Issue TDM load of the whole packed W1 into LDS (wave 0 only) ----
  if (tid < 32) {
    unsigned long long ga = (unsigned long long)(uintptr_t)w1pack;
    // D# group0: count=1, lds_addr=0, global_addr, type=2
    u32x4 g0 = { 1u, 0u,
                 (unsigned)(ga & 0xFFFFFFFFull),
                 (unsigned)((ga >> 32) & 0x1FFFFFFull) | (2u << 30) };
    // D# group1: data_size=8B (code 3); tensor_dim0=32768; tensor_dim1=1;
    // tile_dim0=32768 (1-D 256KB tile); stride dim0 = 32768.
    i32x8 g1 = { (int)(3u << 16),
                 (int)0x80000000u,          // tensor_dim0[15:0] << 16
                 (int)(1u << 16),           // tensor_dim0 hi=0 | tensor_dim1=1
                 (int)0x80000000u,          // tile_dim0 = 32768
                 1,                         // tile_dim1 = 1, tile_dim2 = 0
                 32768, 0, 0 };             // tensor_dim0_stride
    i32x4 g2 = { 0, 0, 0, 0 };
    i32x4 g3 = { 0, 0, 0, 0 };
#if __clang_major__ >= 23
    i32x8 g4 = { 0, 0, 0, 0, 0, 0, 0, 0 };
    __builtin_amdgcn_tensor_load_to_lds(g0, g1, g2, g3, g4, 0);
#else
    __builtin_amdgcn_tensor_load_to_lds(g0, g1, g2, g3, 0);
#endif
  }

  // ---- Preload A operands: this wave's 16x256 feature tile as bf16 ----
  // A layout used here (contiguous k, same order as packed B): lanes 0-15 row M=lane, K = {0..15};
  // lanes 16-31 same rows, K = {16..31}.
  const int t = t0 + wave * 16 + col;
  const float* frow = features + ((size_t)b * Tt + t) * Ee;
  v16bf aop[8];
#pragma unroll
  for (int ks = 0; ks < 8; ++ks) {
    const int e0 = ks * 32 + hi * 16;
    float4 f0 = *reinterpret_cast<const float4*>(frow + e0);
    float4 f1 = *reinterpret_cast<const float4*>(frow + e0 + 4);
    float4 f2 = *reinterpret_cast<const float4*>(frow + e0 + 8);
    float4 f3 = *reinterpret_cast<const float4*>(frow + e0 + 12);
    union { v16bf v; unsigned short s[16]; } u;
    u.s[0]  = f2bf(f0.x); u.s[1]  = f2bf(f0.y); u.s[2]  = f2bf(f0.z); u.s[3]  = f2bf(f0.w);
    u.s[4]  = f2bf(f1.x); u.s[5]  = f2bf(f1.y); u.s[6]  = f2bf(f1.z); u.s[7]  = f2bf(f1.w);
    u.s[8]  = f2bf(f2.x); u.s[9]  = f2bf(f2.y); u.s[10] = f2bf(f2.z); u.s[11] = f2bf(f2.w);
    u.s[12] = f2bf(f3.x); u.s[13] = f2bf(f3.y); u.s[14] = f2bf(f3.z); u.s[15] = f2bf(f3.w);
    aop[ks] = u.v;
  }

  if (tid < 32) __builtin_amdgcn_s_wait_tensorcnt(0);
  __syncthreads();

  const float* phb = ph + (size_t)b * Aa;
  const uint4* bbase = reinterpret_cast<const uint4*>(smem);
  float partial[8] = {0.f, 0.f, 0.f, 0.f, 0.f, 0.f, 0.f, 0.f};

  union BU { v16bf v; uint4 q[2]; };

  for (int nt = 0; nt < 32; ++nt) {
    v8f acc = {0.f, 0.f, 0.f, 0.f, 0.f, 0.f, 0.f, 0.f};
    // Software-pipelined B operand: issue LDS loads for kstep+1 before the
    // WMMA that consumes kstep, so the ds wait overlaps matrix execution.
    BU cur, nxt;
    {
      const int idx0 = ((nt * 8 + 0) * 32 + lane) * 2;
      cur.q[0] = bbase[idx0];
      cur.q[1] = bbase[idx0 + 1];
    }
#pragma unroll
    for (int ks = 0; ks < 8; ++ks) {
      if (ks < 7) {
        const int idx = ((nt * 8 + ks + 1) * 32 + lane) * 2;
        nxt.q[0] = bbase[idx];
        nxt.q[1] = bbase[idx + 1];
      }
      acc = __builtin_amdgcn_wmma_f32_16x16x32_bf16(
          false, aop[ks], false, cur.v, (short)0, acc, false, false);
      cur = nxt;
    }
    // C layout: lane holds (M = r + 8*hi, N = col) for VGPR r; a = nt*16+col
    const int   a   = nt * 16 + col;
    const float phv = phb[a];
    const float wv  = Wv[a];
#pragma unroll
    for (int r = 0; r < 8; ++r) partial[r] += fast_tanh(acc[r] + phv) * wv;
  }

  // Reduce over the 16 lanes of each half-wave (columns of the tile)
#pragma unroll
  for (int m = 1; m < 16; m <<= 1) {
#pragma unroll
    for (int r = 0; r < 8; ++r) partial[r] += __shfl_xor(partial[r], m, 32);
  }

  if (col == 0) {
    const int base = t0 + wave * 16 + hi * 8;
#pragma unroll
    for (int r = 0; r < 8; ++r)
      scores[(size_t)b * Tt + base + r] = partial[r];
  }
}

// ---------------------------------------------------------------------------
// K2a: per-batch softmax stats (max, sum of exp)
// ---------------------------------------------------------------------------
__global__ void softmax_stats_kernel(const float* __restrict__ scores,
                                     float2* __restrict__ stats) {
  __shared__ float red[256];
  const int b = blockIdx.x, tid = threadIdx.x;
  const float* s = scores + (size_t)b * Tt;
  float m = -3.4e38f;
  for (int i = tid; i < Tt; i += 256) m = fmaxf(m, s[i]);
  red[tid] = m; __syncthreads();
  for (int k = 128; k > 0; k >>= 1) {
    if (tid < k) red[tid] = fmaxf(red[tid], red[tid + k]);
    __syncthreads();
  }
  const float bmax = red[0]; __syncthreads();
  float sum = 0.f;
  for (int i = tid; i < Tt; i += 256) sum += __expf(s[i] - bmax);
  red[tid] = sum; __syncthreads();
  for (int k = 128; k > 0; k >>= 1) {
    if (tid < k) red[tid] += red[tid + k];
    __syncthreads();
  }
  if (tid == 0) stats[b] = make_float2(bmax, red[0]);
}

// ---------------------------------------------------------------------------
// K2b: weights out + partial context per (batch, 512-row T chunk)
// ---------------------------------------------------------------------------
__global__ void weights_ctx_kernel(const float* __restrict__ features,
                                   const float* __restrict__ scores,
                                   const float2* __restrict__ stats,
                                   float* __restrict__ out_w,
                                   float* __restrict__ pctx) {
  __shared__ float lw[512];
  const int bid = blockIdx.x;
  const int b = bid >> 3, chunk = bid & 7;
  const int t0 = chunk * 512;
  const int tid = threadIdx.x;
  const float2 st = stats[b];
  const float inv = 1.0f / st.y;
  for (int i = tid; i < 512; i += 256) {
    const float w = __expf(scores[(size_t)b * Tt + t0 + i] - st.x) * inv;
    lw[i] = w;
    out_w[(size_t)b * Tt + t0 + i] = w;
  }
  __syncthreads();
  const int e = tid;                                    // E == blockDim == 256
  const float* f = features + ((size_t)b * Tt + t0) * Ee + e;
  float acc = 0.f;
  for (int i = 0; i < 512; ++i) acc += lw[i] * f[(size_t)i * Ee];
  pctx[(size_t)bid * Ee + e] = acc;
}

// ---------------------------------------------------------------------------
// K2c: reduce the 8 partial contexts per batch
// ---------------------------------------------------------------------------
__global__ void ctx_reduce_kernel(const float* __restrict__ pctx,
                                  float* __restrict__ ctx) {
  const int b = blockIdx.x, e = threadIdx.x;
  float s = 0.f;
#pragma unroll
  for (int c = 0; c < 8; ++c) s += pctx[(size_t)(b * 8 + c) * Ee + e];
  ctx[(size_t)b * Ee + e] = s;
}

// ---------------------------------------------------------------------------
extern "C" void kernel_launch(void* const* d_in, const int* in_sizes, int n_in,
                              void* d_out, int out_size, void* d_ws, size_t ws_size,
                              hipStream_t stream) {
  const float* features = (const float*)d_in[0];
  const float* hidden   = (const float*)d_in[1];
  const float* W1 = (const float*)d_in[2];
  const float* b1 = (const float*)d_in[3];
  const float* W2 = (const float*)d_in[4];
  const float* b2 = (const float*)d_in[5];
  const float* Wv = (const float*)d_in[6];
  // d_in[7] = bv: adding a constant to all scores is softmax-invariant -> unused.

  char* ws = (char*)d_ws;
  float*          ph     = (float*)(ws);                  // 64*512*4   = 128KB
  unsigned short* w1p    = (unsigned short*)(ws + 131072);// 256*512*2  = 256KB
  float*          scores = (float*)(ws + 393216);         // 64*4096*4  = 1MB
  float2*         stats  = (float2*)(ws + 1441792);       // 512B
  float*          pctx   = (float*)(ws + 1442304);        // 64*8*256*4 = 512KB

  float* ctx_out = (float*)d_out;                         // (B,E)
  float* w_out   = (float*)d_out + Bb * Ee;               // (B,T,1)

  pack_w1_kernel   <<<32, 256, 0, stream>>>(W1, w1p);
  proj_h_kernel    <<<Bb, 256, 0, stream>>>(hidden, W2, b1, b2, ph);
  attn_score_kernel<<<Bb * (Tt / 128), 256, 262144, stream>>>(features, w1p, ph, Wv, scores);
  softmax_stats_kernel<<<Bb, 256, 0, stream>>>(scores, stats);
  weights_ctx_kernel  <<<Bb * 8, 256, 0, stream>>>(features, scores, stats, w_out, pctx);
  ctx_reduce_kernel   <<<Bb, 256, 0, stream>>>(pctx, ctx_out);
}
